// SelfMatcher1_53120155517071
// MI455X (gfx1250) — matrix operand order repaired and checked by reading, not compiled, not run on hardware
//
#include <hip/hip_runtime.h>
#include <math.h>

typedef __attribute__((ext_vector_type(16))) _Float16 v16h;
typedef __attribute__((ext_vector_type(8)))  float    v8f;
typedef unsigned int u32x4 __attribute__((ext_vector_type(4)));
typedef int          i32x8 __attribute__((ext_vector_type(8)));
typedef int          i32x4 __attribute__((ext_vector_type(4)));

#define L_SEQ 2048
#define BATCH 16
#define DIM   128
#define G3    384   // 3*DIM

// ---------------- WMMA fragment helpers (CDNA5 16x16x32 f16, wave32) ----------------

__device__ __forceinline__ v8f wmma_f16(v16h a, v16h b, v8f c) {
  return __builtin_amdgcn_wmma_f32_16x16x32_f16(false, a, false, b, (short)0, c, false, false);
}

// A 16x32 f16 fragment from row-major [.. x ld], base at (row0,k0).
// lane holds row m=lane&15; lanes 0-15: K=0..15; lanes 16-31: K=16..31.
__device__ __forceinline__ v16h load_a_frag(const _Float16* base, int ld, int lane) {
  int m  = lane & 15;
  int kb = (lane & 16) ? 16 : 0;
  const _Float16* p = base + m * ld + kb;
  v16h a;
#pragma unroll
  for (int h = 0; h < 8; ++h) a[h] = p[h];
#pragma unroll
  for (int h = 0; h < 8; ++h) a[h + 8] = p[h + 8];
  return a;
}

// B 32x16 f16 fragment, B row-major [K x ld] at (k0,col0).
// lane holds col n=lane&15; lanes 0-15: K=0..15; lanes 16-31: K=16..31.
__device__ __forceinline__ v16h load_b_frag(const _Float16* base, int ld, int lane) {
  int n  = lane & 15;
  int kb = (lane & 16) ? 16 : 0;
  v16h b;
#pragma unroll
  for (int h = 0; h < 16; ++h) b[h] = base[(kb + h) * ld + n];
  return b;
}

// B 32x16 fragment where B[k][n] = M[n][k] (M row-major stride ld): contiguous per lane.
__device__ __forceinline__ v16h load_b_frag_T(const _Float16* base, int ld, int lane) {
  int n  = lane & 15;
  int kb = (lane & 16) ? 16 : 0;
  const _Float16* p = base + n * ld + kb;
  v16h b;
#pragma unroll
  for (int h = 0; h < 16; ++h) b[h] = p[h];
  return b;
}

// ---------------- Tensor Data Mover: 1-D async copy global(f16) -> LDS ----------------
// D# group0: count=1 | lds_addr | global_addr | type=2.  group1: data_size=2B, 1-D tile.

__device__ __forceinline__ void tdm_load_f16_1d(unsigned lds_byte_addr,
                                                const _Float16* src, unsigned nelem) {
  unsigned long long ga = (unsigned long long)(uintptr_t)src;
  u32x4 g0;
  g0[0] = 1u;                                         // count=1 (valid), user mode
  g0[1] = lds_byte_addr;                              // LDS destination (bytes)
  g0[2] = (unsigned)ga;                               // global_addr[31:0]
  g0[3] = (unsigned)((ga >> 32) & 0x1FFFFFFu) | (2u << 30);  // addr[56:32] | type=2
  i32x8 g1;
  g1[0] = (int)(1u << 16);                            // workgroup_mask=0, data_size=1 (2B)
  g1[1] = (int)(nelem << 16);                         // tensor_dim0[15:0]
  g1[2] = (int)((nelem >> 16) | (1u << 16));          // tensor_dim0[31:16] | tensor_dim1=1
  g1[3] = (int)(nelem << 16);                         // tile_dim0 = nelem
  g1[4] = 1;                                          // tile_dim1=1, tile_dim2=0
  g1[5] = (int)nelem;                                 // tensor_dim0_stride[31:0]
  g1[6] = 0;
  g1[7] = 0;
  i32x4 z4 = (i32x4){0, 0, 0, 0};
  i32x8 z8 = (i32x8){0, 0, 0, 0, 0, 0, 0, 0};
  __builtin_amdgcn_tensor_load_to_lds(g0, g1, z4, z4, z8, 0);  // clang-23 6-arg form
}

// ---------------- Kernel 1: f16 copy of v (transposed to [B][L][D]) + row norms ----------------

__global__ void prep_kernel(const float* __restrict__ v, _Float16* __restrict__ vh,
                            float* __restrict__ sq) {
  int row = blockIdx.x;            // row = i*BATCH + b  (v is [L][B][D])
  int i = row / BATCH, b = row % BATCH;
  int d = threadIdx.x;
  float x = v[(size_t)row * DIM + d];
  vh[((size_t)b * L_SEQ + i) * DIM + d] = (_Float16)x;
  float s = x * x;
#pragma unroll
  for (int o = 16; o > 0; o >>= 1) s += __shfl_xor(s, o, 32);
  __shared__ float part[4];
  if ((threadIdx.x & 31) == 0) part[threadIdx.x >> 5] = s;
  __syncthreads();
  if (threadIdx.x == 0) sq[b * L_SEQ + i] = part[0] + part[1] + part[2] + part[3];
}

// ---------------- Kernel 2: weight transpose + f16 convert ----------------

__global__ void wconv_kernel(const float* __restrict__ Wih, const float* __restrict__ Whh,
                             _Float16* __restrict__ WihT, _Float16* __restrict__ WhhT) {
  int idx = blockIdx.x * blockDim.x + threadIdx.x;   // over G3*DIM
  if (idx >= G3 * DIM) return;
  int n = idx / DIM, k = idx % DIM;
  WihT[k * G3 + n] = (_Float16)Wih[n * DIM + k];
  WhhT[k * G3 + n] = (_Float16)Whh[n * DIM + k];
}

// ---------------- Kernel 3: attention (flash style, analytic row-max) ----------------
// One wave per (b, 16-row query tile). S<=0 with S[i,i]=0 -> exp(S) is already max-subtracted.

__global__ void __launch_bounds__(32, 1)
attn_kernel(const _Float16* __restrict__ vh, const float* __restrict__ sq,
            _Float16* __restrict__ Ch) {
  int it = blockIdx.x;
  int b  = blockIdx.y;
  int lane = threadIdx.x & 31;
  int i0 = it * 16;
  int nn = lane & 15;
  int mbase = (lane & 16) ? 8 : 0;

  const _Float16* vb = vh + (size_t)b * L_SEQ * DIM;
  const float* sqb = sq + b * L_SEQ;

  v16h aq[4];
#pragma unroll
  for (int kc = 0; kc < 4; ++kc) aq[kc] = load_a_frag(vb + (size_t)i0 * DIM + kc * 32, DIM, lane);

  float si[8];
#pragma unroll
  for (int e = 0; e < 8; ++e) si[e] = sqb[i0 + mbase + e];

  v8f acc[8];
#pragma unroll
  for (int nt = 0; nt < 8; ++nt) acc[nt] = (v8f){};
  float denom[8];
#pragma unroll
  for (int e = 0; e < 8; ++e) denom[e] = 0.f;

  __shared__ _Float16 pl[16 * 32];   // P tile staging: D-layout -> A-layout

  for (int jt = 0; jt < L_SEQ / 32; ++jt) {
    int j0 = jt * 32;
#pragma unroll
    for (int hh = 0; hh < 2; ++hh) {
      int jj0 = j0 + hh * 16;
      // preload all 4 K-chunk B fragments so loads clause together and
      // the 4 WMMAs issue back-to-back behind a single wait
      v16h bq[4];
#pragma unroll
      for (int kc = 0; kc < 4; ++kc)
        bq[kc] = load_b_frag_T(vb + (size_t)jj0 * DIM + kc * 32, DIM, lane);
      float sqj = sqb[jj0 + nn];
      v8f s = (v8f){};
#pragma unroll
      for (int kc = 0; kc < 4; ++kc) s = wmma_f16(aq[kc], bq[kc], s);
#pragma unroll
      for (int e = 0; e < 8; ++e) {
        float p = __expf(2.0f * s[e] - sqj - si[e]);   // logit <= 0, stable
        denom[e] += p;
        pl[(mbase + e) * 32 + hh * 16 + nn] = (_Float16)p;
      }
    }
    __syncthreads();
    v16h pa = load_a_frag(pl, 32, lane);
#pragma unroll
    for (int nt = 0; nt < 8; ++nt)
      acc[nt] = wmma_f16(pa, load_b_frag(vb + (size_t)j0 * DIM + nt * 16, DIM, lane), acc[nt]);
    __syncthreads();
  }

#pragma unroll
  for (int e = 0; e < 8; ++e) {
    float d = denom[e];
    d += __shfl_xor(d, 1, 32);
    d += __shfl_xor(d, 2, 32);
    d += __shfl_xor(d, 4, 32);
    d += __shfl_xor(d, 8, 32);
    denom[e] = 1.0f / d;
  }

#pragma unroll
  for (int nt = 0; nt < 8; ++nt)
#pragma unroll
    for (int e = 0; e < 8; ++e)
      Ch[(((size_t)(i0 + mbase + e)) * BATCH + b) * DIM + nt * 16 + nn] =
          (_Float16)(acc[nt][e] * denom[e]);
}

// ---------------- Kernel 4: Gi = Ch @ W_ih^T + b_ih, stored f16 [L*B][384] ----------------

__global__ void __launch_bounds__(32, 1)
gigemm_kernel(const _Float16* __restrict__ Ch, const _Float16* __restrict__ WihT,
              const float* __restrict__ bih, _Float16* __restrict__ Gi) {
  int lane = threadIdx.x & 31;
  int m0 = blockIdx.x * 16;
  int n0 = blockIdx.y * 16;
  int nn = lane & 15;
  int mbase = (lane & 16) ? 8 : 0;

  v8f acc = (v8f){};
#pragma unroll
  for (int kc = 0; kc < 4; ++kc)
    acc = wmma_f16(load_a_frag(Ch + (size_t)m0 * DIM + kc * 32, DIM, lane),
                   load_b_frag(WihT + (size_t)kc * 32 * G3 + n0, G3, lane), acc);
  float bias = bih[n0 + nn];
#pragma unroll
  for (int e = 0; e < 8; ++e)
    Gi[((size_t)(m0 + mbase + e)) * G3 + n0 + nn] = (_Float16)(acc[e] + bias);
}

// ---------------- Kernel 5: sequential GRU scan, TDM double-buffered gates ----------------
// 8 waves; W_hh^T fragments in registers; h in registers+LDS(f16); Gi[t] streamed by TDM.

__global__ void __launch_bounds__(256, 1)
gru_kernel(const _Float16* __restrict__ Gi, const _Float16* __restrict__ WhhT,
           const float* __restrict__ bhh, const float* __restrict__ h0,
           float* __restrict__ out) {
  int tid  = threadIdx.x;
  int lane = tid & 31;
  int wave = tid >> 5;
  int nn = lane & 15;
  int mbase = (lane & 16) ? 8 : 0;

  __shared__ _Float16 hf16[BATCH * DIM];        // 4 KB: h as WMMA A operand
  __shared__ float    gh[BATCH * G3];           // 24 KB: recurrent preactivations
  __shared__ _Float16 gi_buf[2][BATCH * G3];    // 24 KB: TDM double buffer

  // W_hh^T fragments: wave w owns n-tiles 3w..3w+2 (12 v16h = 96 VGPRs)
  v16h wb[3][4];
  float bias[3];
#pragma unroll
  for (int j = 0; j < 3; ++j) {
    int n0 = (wave * 3 + j) * 16;
    bias[j] = bhh[n0 + nn];
#pragma unroll
    for (int kc = 0; kc < 4; ++kc)
      wb[j][kc] = load_b_frag(WhhT + (size_t)kc * 32 * G3 + n0, G3, lane);
  }

  // h lives in registers: thread owns elements e = tid + k*256
  float hreg[8];
#pragma unroll
  for (int k = 0; k < 8; ++k) {
    int e = tid + k * 256;
    float x = h0[e];
    hreg[k] = x;
    hf16[e] = (_Float16)x;
  }

  // kick off TDM load of Gi[0]
  if (wave == 0)
    tdm_load_f16_1d((unsigned)(uintptr_t)&gi_buf[0][0], Gi, BATCH * G3);
  __syncthreads();

  for (int t = 0; t < L_SEQ; ++t) {
    int cur = t & 1;
    // prefetch Gi[t+1] into the other buffer while we do the recurrent GEMM
    if (wave == 0 && t + 1 < L_SEQ)
      tdm_load_f16_1d((unsigned)(uintptr_t)&gi_buf[1 - cur][0],
                      Gi + (size_t)(t + 1) * BATCH * G3, BATCH * G3);

    // gh = h @ W_hh^T + b_hh
    v16h ha[4];
#pragma unroll
    for (int kc = 0; kc < 4; ++kc) ha[kc] = load_a_frag(hf16 + kc * 32, DIM, lane);
#pragma unroll
    for (int j = 0; j < 3; ++j) {
      v8f acc = (v8f){};
#pragma unroll
      for (int kc = 0; kc < 4; ++kc) acc = wmma_f16(ha[kc], wb[j][kc], acc);
      int n0 = (wave * 3 + j) * 16;
#pragma unroll
      for (int e = 0; e < 8; ++e)
        gh[(mbase + e) * G3 + n0 + nn] = acc[e] + bias[j];
    }

    // tensor ops complete in order: <=1 outstanding means Gi[t] has landed
    if (wave == 0) {
      if (t + 1 < L_SEQ) __builtin_amdgcn_s_wait_tensorcnt(1);
      else               __builtin_amdgcn_s_wait_tensorcnt(0);
    }
    __syncthreads();

    // elementwise GRU update: 2048 elements over 256 threads
    const _Float16* gi_t = &gi_buf[cur][0];
    float* out_t = out + (size_t)t * BATCH * DIM;
#pragma unroll
    for (int k = 0; k < 8; ++k) {
      int e  = tid + k * 256;
      int bb = e >> 7, d = e & 127;
      float gir = (float)gi_t[bb * G3 + d];
      float giz = (float)gi_t[bb * G3 + 128 + d];
      float gin = (float)gi_t[bb * G3 + 256 + d];
      float ghr = gh[bb * G3 + d];
      float ghz = gh[bb * G3 + 128 + d];
      float ghn = gh[bb * G3 + 256 + d];
      float r = 1.0f / (1.0f + __expf(-(gir + ghr)));
      float z = 1.0f / (1.0f + __expf(-(giz + ghz)));
      float n = tanhf(gin + r * ghn);
      float hnew = (1.0f - z) * n + z * hreg[k];
      hreg[k] = hnew;
      out_t[e] = hnew;
      hf16[e] = (_Float16)hnew;
    }
    __syncthreads();
  }
}

// ---------------- Launch ----------------

extern "C" void kernel_launch(void* const* d_in, const int* in_sizes, int n_in,
                              void* d_out, int out_size, void* d_ws, size_t ws_size,
                              hipStream_t stream) {
  const float* v   = (const float*)d_in[0];
  const float* Wih = (const float*)d_in[1];
  const float* Whh = (const float*)d_in[2];
  const float* bih = (const float*)d_in[3];
  const float* bhh = (const float*)d_in[4];
  const float* h0  = (const float*)d_in[5];
  float* out = (float*)d_out;

  char* w = (char*)d_ws;
  _Float16* vh = (_Float16*)w;  w += (size_t)BATCH * L_SEQ * DIM * sizeof(_Float16);  // 8 MB
  _Float16* Ch = (_Float16*)w;  w += (size_t)L_SEQ * BATCH * DIM * sizeof(_Float16);  // 8 MB
  float* sq = (float*)w;        w += (size_t)BATCH * L_SEQ * sizeof(float);           // 128 KB
  _Float16* WihT = (_Float16*)w; w += (size_t)G3 * DIM * sizeof(_Float16);            // 96 KB
  _Float16* WhhT = (_Float16*)w; w += (size_t)G3 * DIM * sizeof(_Float16);            // 96 KB
  w = (char*)(((uintptr_t)w + 255) & ~(uintptr_t)255);
  _Float16* Gi = (_Float16*)w;  w += (size_t)L_SEQ * BATCH * G3 * sizeof(_Float16);   // 24 MB

  prep_kernel<<<L_SEQ * BATCH, DIM, 0, stream>>>(v, vh, sq);
  wconv_kernel<<<(G3 * DIM + 255) / 256, 256, 0, stream>>>(Wih, Whh, WihT, WhhT);
  attn_kernel<<<dim3(L_SEQ / 16, BATCH), 32, 0, stream>>>(vh, sq, Ch);
  gigemm_kernel<<<dim3(L_SEQ * BATCH / 16, G3 / 16), 32, 0, stream>>>(Ch, WihT, bih, Gi);
  gru_kernel<<<1, 256, 0, stream>>>(Gi, WhhT, bhh, h0, out);
}
